// CrissCrossSelfAttention_78898549227846
// MI455X (gfx1250) — matrix operand order repaired and checked by reading, not compiled, not run on hardware
//
#include <hip/hip_runtime.h>

typedef __attribute__((ext_vector_type(16))) __bf16 v16bf;
typedef __attribute__((ext_vector_type(8)))  __bf16 v8bf;
typedef __attribute__((ext_vector_type(8)))  float  v8f;

#define NHEADS 8
#define CH     512
#define HDIM   128
#define WDIM   128
#define HW     (HDIM * WDIM)
#define NBATCH 4

// ---------------- WMMA fragment loaders (bf16 16x16x32), wide LDS loads ----------
// A 16x32: lane row m=lane&15; elem e -> K = ((e>>3)<<4) + half*8 + (e&7)
//   => elems 0..7 contiguous at K=half*8, elems 8..15 contiguous at K=16+half*8
// B 32x16: lane col n=lane&15; elem e -> K = ((e>>3)<<4) + half*8 + (e&7)  (same as A)
// C/D 16x16 f32: VGPR g, lane l: M=(l>>4)*8+g, N=l&15

__device__ __forceinline__ v16bf loadFragA16(const __bf16* rowBase, int half) {
  v8bf lo = *(const v8bf*)(rowBase + half * 8);
  v8bf hi = *(const v8bf*)(rowBase + 16 + half * 8);
  return __builtin_shufflevector(lo, hi, 0, 1, 2, 3, 4, 5, 6, 7,
                                 8, 9, 10, 11, 12, 13, 14, 15);
}
__device__ __forceinline__ v16bf loadFragB16(const __bf16* colBase, int half) {
  v8bf lo = *(const v8bf*)(colBase + half * 8);
  v8bf hi = *(const v8bf*)(colBase + 16 + half * 8);
  return __builtin_shufflevector(lo, hi, 0, 1, 2, 3, 4, 5, 6, 7,
                                 8, 9, 10, 11, 12, 13, 14, 15);
}

__device__ __forceinline__ __bf16 toBf(float x)  { return (__bf16)x; }
__device__ __forceinline__ __bf16 toBf(__bf16 x) { return x; }
__device__ __forceinline__ void storeOut(float*  p, float v) { *p = v; }
__device__ __forceinline__ void storeOut(__bf16* p, float v) { *p = (__bf16)v; }

// ---------------- Generic channel GEMM: Out[b] = Wm (Cout x Cin) * X[b] (Cin x Nn) --
// block tile 128x128, 256 threads = 8 waves in 2(M) x 4(N); wave tile 64x32.
// Double-buffered LDS pipeline: global loads for tile kt+1 overlap WMMAs of tile kt.
template <typename TB, typename TOut>
__global__ __launch_bounds__(256)
void gemm_wx(const float* __restrict__ Wm, const TB* __restrict__ X,
             TOut* __restrict__ Out, int Cin, int Cout, int Nn) {
  __shared__ __align__(16) __bf16 As[2][128 * 32];   // [m][k], ld=32
  __shared__ __align__(16) __bf16 Bs[2][128 * 32];   // [n][k] (transposed), ld=32
  const int t = threadIdx.x;
  const int lane = t & 31, wid = t >> 5;
  const int half = lane >> 4, l15 = lane & 15;
  const int waveM = wid >> 2, waveN = wid & 3;
  const int m0 = blockIdx.y * 128;
  const int n0 = blockIdx.x * 128;
  const size_t bX = (size_t)blockIdx.z * (size_t)Cin * Nn;
  const size_t bO = (size_t)blockIdx.z * (size_t)Cout * Nn;

  v8f acc[4][2];
#pragma unroll
  for (int mi = 0; mi < 4; ++mi)
#pragma unroll
    for (int ni = 0; ni < 2; ++ni)
#pragma unroll
      for (int g = 0; g < 8; ++g) acc[mi][ni][g] = 0.0f;

  // staging assignments (per thread, fixed across tiles)
  const int ar = t >> 1, ac0 = (t & 1) * 16;      // A: row 0..127, 16 cols
  const int br = t >> 3, bc0 = (t & 7) * 16;      // B: k-row 0..31, 16 n-cols
  const float* aSrc = &Wm[(size_t)(m0 + ar) * Cin + ac0];
  const TB*    bSrc = &X[bX + (size_t)br * Nn + n0 + bc0];

  float ta[16];
  TB    tb[16];
  const int kTiles = Cin >> 5;

  // ---- preload + store tile 0 ----
#pragma unroll
  for (int i = 0; i < 16; i += 4) {
    float4 f4 = *(const float4*)(aSrc + i);
    ta[i + 0] = f4.x; ta[i + 1] = f4.y; ta[i + 2] = f4.z; ta[i + 3] = f4.w;
  }
#pragma unroll
  for (int i = 0; i < 16; ++i) tb[i] = bSrc[i];
  {
    union { __bf16 h[16]; v8bf v[2]; } u;
#pragma unroll
    for (int i = 0; i < 16; ++i) u.h[i] = toBf(ta[i]);
    *(v8bf*)&As[0][ar * 32 + ac0]     = u.v[0];
    *(v8bf*)&As[0][ar * 32 + ac0 + 8] = u.v[1];
#pragma unroll
    for (int i = 0; i < 16; ++i) Bs[0][(bc0 + i) * 32 + br] = toBf(tb[i]);
  }

  for (int kt = 0; kt < kTiles; ++kt) {
    const int cur = kt & 1;
    const bool hasNext = (kt + 1 < kTiles);
    __syncthreads();   // lds[cur] ready for all; lds[1-cur] free to overwrite

    // issue next tile's global loads (latency hidden under the WMMAs below)
    if (hasNext) {
      const float* an = aSrc + (size_t)(kt + 1) * 32;
      const TB*    bn = bSrc + (size_t)(kt + 1) * 32 * Nn;
#pragma unroll
      for (int i = 0; i < 16; i += 4) {
        float4 f4 = *(const float4*)(an + i);
        ta[i + 0] = f4.x; ta[i + 1] = f4.y; ta[i + 2] = f4.z; ta[i + 3] = f4.w;
      }
#pragma unroll
      for (int i = 0; i < 16; ++i) tb[i] = bn[i];
      if (kt + 2 < kTiles) {
        __builtin_prefetch(aSrc + (size_t)(kt + 2) * 32, 0, 0);
        __builtin_prefetch(bSrc + (size_t)(kt + 2) * 32 * Nn, 0, 0);
      }
    }

    // compute from lds[cur]
    {
      v16bf af[4], bfr[2];
#pragma unroll
      for (int mi = 0; mi < 4; ++mi)
        af[mi] = loadFragA16(&As[cur][(waveM * 64 + mi * 16 + l15) * 32], half);
#pragma unroll
      for (int ni = 0; ni < 2; ++ni)
        bfr[ni] = loadFragB16(&Bs[cur][(waveN * 32 + ni * 16 + l15) * 32], half);
#pragma unroll
      for (int mi = 0; mi < 4; ++mi)
#pragma unroll
        for (int ni = 0; ni < 2; ++ni)
          acc[mi][ni] = __builtin_amdgcn_wmma_f32_16x16x32_bf16(
              false, af[mi], false, bfr[ni], (short)0, acc[mi][ni], false, false);
    }

    // store next tile into the other buffer (no barrier needed: disjoint buffer)
    if (hasNext) {
      const int nxt = 1 - cur;
      union { __bf16 h[16]; v8bf v[2]; } u;
#pragma unroll
      for (int i = 0; i < 16; ++i) u.h[i] = toBf(ta[i]);
      *(v8bf*)&As[nxt][ar * 32 + ac0]     = u.v[0];
      *(v8bf*)&As[nxt][ar * 32 + ac0 + 8] = u.v[1];
#pragma unroll
      for (int i = 0; i < 16; ++i) Bs[nxt][(bc0 + i) * 32 + br] = toBf(tb[i]);
    }
  }

#pragma unroll
  for (int mi = 0; mi < 4; ++mi)
#pragma unroll
    for (int ni = 0; ni < 2; ++ni)
#pragma unroll
      for (int g = 0; g < 8; ++g) {
        int row = m0 + waveM * 64 + mi * 16 + half * 8 + g;
        int col = n0 + waveN * 32 + ni * 16 + l15;
        storeOut(&Out[bO + (size_t)row * Nn + col], acc[mi][ni][g]);
      }
}

// ---------------- Axial attention: one workgroup per (b, head, line, axis) ---------
// mode 0: vertical (attend along H, fixed w); mode 1: horizontal (attend along W, fixed h)
__global__ __launch_bounds__(256)
void cc_attn(const __bf16* __restrict__ q, const __bf16* __restrict__ k,
             const __bf16* __restrict__ v, __bf16* __restrict__ cat) {
  __shared__ __align__(16) char smem[48 * 1024];
  __bf16* qS = (__bf16*)smem;               // [p][cc], ld=64  (16KB)
  __bf16* kS = (__bf16*)(smem + 16384);     // [p][cc], ld=64  (16KB)
  __bf16* vS = (__bf16*)(smem + 32768);     // [cc][p], ld=128 (16KB)
  __bf16* Sb = (__bf16*)smem;               // [m][k] 128x128 bf16, overlays qS+kS

  const int t = threadIdx.x, lane = t & 31, wid = t >> 5;
  const int half = lane >> 4, l15 = lane & 15;
  const int fixed = blockIdx.x;
  const int head = blockIdx.y & 7, b = blockIdx.y >> 3;
  const int mode = blockIdx.z;
  const size_t chanBase = ((size_t)b * CH + head * 64) * HW;
  const int posStride = (mode == 0) ? WDIM : 1;
  const int fixedOff  = (mode == 0) ? fixed : fixed * WDIM;

  // load q/k/v slices: thread owns channel cc = t>>2, 32 contiguous positions
  {
    const int cc = t >> 2;
    const int p0 = (t & 3) * 32;
    const size_t gBase = chanBase + (size_t)cc * HW + fixedOff;
#pragma unroll
    for (int i = 0; i < 32; ++i) {
      const int p = p0 + i;
      const size_t g = gBase + (size_t)p * posStride;
      qS[p * 64 + cc]  = q[g];
      kS[p * 64 + cc]  = k[g];
      vS[cc * 128 + p] = v[g];
    }
  }
  __syncthreads();

  // Phase 1: S = Q^T K (128x128, K=64); wave wid computes rows [16*wid, 16*wid+16)
  const int mrow = wid * 16;
  v8f acc[8];
#pragma unroll
  for (int nt = 0; nt < 8; ++nt)
#pragma unroll
    for (int g = 0; g < 8; ++g) acc[nt][g] = 0.0f;

#pragma unroll
  for (int ks_ = 0; ks_ < 2; ++ks_) {
    v16bf a = loadFragA16(&qS[(mrow + l15) * 64 + ks_ * 32], half);       // A[m][cc]
#pragma unroll
    for (int nt = 0; nt < 8; ++nt) {
      v16bf bb = loadFragB16(&kS[(nt * 16 + l15) * 64 + ks_ * 32], half); // B[cc][n]
      acc[nt] = __builtin_amdgcn_wmma_f32_16x16x32_bf16(
          false, a, false, bb, (short)0, acc[nt], false, false);
    }
  }
  __syncthreads();   // all waves done reading qS/kS before Sb overlays them

  // Phase 2: register softmax along the full row (cols across 16-lane half x 8 tiles)
#pragma unroll
  for (int g = 0; g < 8; ++g) {
    float m = -3.402823466e38f;
#pragma unroll
    for (int nt = 0; nt < 8; ++nt) m = fmaxf(m, acc[nt][g]);
#pragma unroll
    for (int off = 1; off <= 8; off <<= 1) m = fmaxf(m, __shfl_xor(m, off, 32));
    float s = 0.0f;
#pragma unroll
    for (int nt = 0; nt < 8; ++nt) {
      float e = __expf(acc[nt][g] - m);
      acc[nt][g] = e; s += e;
    }
#pragma unroll
    for (int off = 1; off <= 8; off <<= 1) s += __shfl_xor(s, off, 32);
    float inv = 1.0f / s;
    int r = mrow + half * 8 + g;
#pragma unroll
    for (int nt = 0; nt < 8; ++nt)
      Sb[r * 128 + nt * 16 + l15] = (__bf16)(acc[nt][g] * inv);
  }
  __syncthreads();

  // Phase 3: Out^T (128 x 64) = attn (128x128) * V^T (128x64)
  v8f acc2[4];
#pragma unroll
  for (int nt = 0; nt < 4; ++nt)
#pragma unroll
    for (int g = 0; g < 8; ++g) acc2[nt][g] = 0.0f;

#pragma unroll
  for (int ks_ = 0; ks_ < 4; ++ks_) {
    v16bf a = loadFragA16(&Sb[(mrow + l15) * 128 + ks_ * 32], half);       // A[m][l]
#pragma unroll
    for (int nt = 0; nt < 4; ++nt) {
      v16bf bb = loadFragB16(&vS[(nt * 16 + l15) * 128 + ks_ * 32], half); // B[l][cc]
      acc2[nt] = __builtin_amdgcn_wmma_f32_16x16x32_bf16(
          false, a, false, bb, (short)0, acc2[nt], false, false);
    }
  }

  // scatter into concatenated (B, 2C, H, W) buffer
#pragma unroll
  for (int nt = 0; nt < 4; ++nt)
#pragma unroll
    for (int g = 0; g < 8; ++g) {
      int p  = mrow + half * 8 + g;         // position along attended axis
      int cc = nt * 16 + l15;
      int ch = mode * CH + head * 64 + cc;
      int h_ = (mode == 0) ? p : fixed;
      int w_ = (mode == 0) ? fixed : p;
      cat[(((size_t)b * (2 * CH) + ch) * HDIM + h_) * WDIM + w_] = (__bf16)acc2[nt][g];
    }
}

// ---------------- host launch ----------------
extern "C" void kernel_launch(void* const* d_in, const int* in_sizes, int n_in,
                              void* d_out, int out_size, void* d_ws, size_t ws_size,
                              hipStream_t stream) {
  (void)in_sizes; (void)n_in; (void)out_size; (void)ws_size;
  const float* x  = (const float*)d_in[0];
  const float* Wq = (const float*)d_in[1];
  const float* Wk = (const float*)d_in[2];
  const float* Wv = (const float*)d_in[3];
  const float* Wo = (const float*)d_in[4];
  float* out = (float*)d_out;

  const size_t qkvBytes = (size_t)NBATCH * CH * HW * sizeof(__bf16);   // 64 MiB each
  char* ws = (char*)d_ws;
  __bf16* qb   = (__bf16*)(ws);
  __bf16* kb   = (__bf16*)(ws + qkvBytes);
  __bf16* vb   = (__bf16*)(ws + 2 * qkvBytes);
  __bf16* catb = (__bf16*)(ws + 3 * qkvBytes);                         // 128 MiB

  dim3 blk(256);
  dim3 gProj(HW / 128, CH / 128, NBATCH);
  gemm_wx<float, __bf16><<<gProj, blk, 0, stream>>>(Wq, x, qb, CH, CH, HW);
  gemm_wx<float, __bf16><<<gProj, blk, 0, stream>>>(Wk, x, kb, CH, CH, HW);
  gemm_wx<float, __bf16><<<gProj, blk, 0, stream>>>(Wv, x, vb, CH, CH, HW);

  dim3 gAtt(HDIM, NHEADS * NBATCH, 2);   // (line, b*head, axis)
  cc_attn<<<gAtt, blk, 0, stream>>>(qb, kb, vb, catb);

  dim3 gOut(HW / 128, CH / 128, NBATCH);
  gemm_wx<__bf16, float><<<gOut, blk, 0, stream>>>(Wo, catb, out, 2 * CH, CH, HW);
}
